// GraphTransformer_24223615549762
// MI455X (gfx1250) — matrix operand order repaired and checked by reading, not compiled, not run on hardware
//
#include <hip/hip_runtime.h>
#include <cstdint>
#include <cstddef>

// ---------------------------------------------------------------------------
// GraphTransformer for MI455X (gfx1250, wave32, WMMA).
//   h0 = adj @ x   -> bf16 hi/lo split WMMA GEMM; adj streamed from HBM once;
//                     x-panels staged to LDS with async loads (double-buffered,
//                     ASYNCcnt), M=32/wave so each LDS B read feeds 6 WMMAs.
//   post-pipeline  -> fused per-16-row WG: QKV, head-softmax, Wo+LN, FFN+LN,
//                     classifier; all GEMMs via v_wmma_f32_16x16x32_bf16.
// ---------------------------------------------------------------------------

#define NN 16384    // nodes
#define ND 128      // node dim
#define KSPLIT 4    // K-partitions of the big GEMM (occupancy)
#define KCHUNK 64   // K elements staged to LDS per stage (2 WMMA k-steps)
#define PSTRIDE 136 // ushorts per LDS panel row: 64 hi + 64 lo + 8 pad (bank-safe)

typedef __attribute__((ext_vector_type(16))) __bf16       v16bf;
typedef __attribute__((ext_vector_type(8)))  float        v8f;
typedef __attribute__((ext_vector_type(8)))  unsigned int v8u;

#if __has_builtin(__builtin_amdgcn_cvt_pk_bf16_f32)
#define HAVE_CVT_PK_BF16 1
#endif

// ---- bf16 helpers (RNE split: f ~= hi + lo, residual error ~2^-16) ----------
__device__ __forceinline__ unsigned bf16_rne_bits(float f) {
  unsigned u = __builtin_bit_cast(unsigned, f);
  unsigned r = u + 0x7fffu + ((u >> 16) & 1u);
  return r >> 16;
}
__device__ __forceinline__ void split1(float f, unsigned short& h, unsigned short& l) {
  unsigned hb = bf16_rne_bits(f);
  float hf = __builtin_bit_cast(float, hb << 16);
  h = (unsigned short)hb;
  l = (unsigned short)bf16_rne_bits(f - hf);
}
// 16 f32 -> packed hi/lo v16bf (VGPR j = element pair {2j, 2j+1})
__device__ __forceinline__ void split16(const float* g, v16bf& H, v16bf& L) {
  v8u hu, lu;
#pragma unroll
  for (int i = 0; i < 8; ++i) {
    float f0 = g[2 * i], f1 = g[2 * i + 1];
#ifdef HAVE_CVT_PK_BF16
    auto ph = __builtin_amdgcn_cvt_pk_bf16_f32(f0, f1);
    unsigned hb = __builtin_bit_cast(unsigned, ph);
#else
    unsigned hb = bf16_rne_bits(f0) | (bf16_rne_bits(f1) << 16);
#endif
    float r0 = f0 - __builtin_bit_cast(float, hb << 16);
    float r1 = f1 - __builtin_bit_cast(float, hb & 0xffff0000u);
#ifdef HAVE_CVT_PK_BF16
    auto pl = __builtin_amdgcn_cvt_pk_bf16_f32(r0, r1);
    unsigned lb = __builtin_bit_cast(unsigned, pl);
#else
    unsigned lb = bf16_rne_bits(r0) | (bf16_rne_bits(r1) << 16);
#endif
    hu[i] = hb;
    lu[i] = lb;
  }
  H = __builtin_bit_cast(v16bf, hu);
  L = __builtin_bit_cast(v16bf, lu);
}
__device__ __forceinline__ v8f v8f_zero() {
  v8f z = {0.f, 0.f, 0.f, 0.f, 0.f, 0.f, 0.f, 0.f};
  return z;
}
__device__ __forceinline__ v8f wmma_bf16(v16bf a, v16bf b, v8f c) {
  return __builtin_amdgcn_wmma_f32_16x16x32_bf16(false, a, false, b, (short)0, c,
                                                 false, false);
}

// A-operand fetch (16 f32, contiguous K order):
// lane<16 row=lane K{0..15}; lane>=16 row=lane-16 K{16..31}.
__device__ __forceinline__ void fetchA(const float* row_ptr, int koff, float4& a0,
                                       float4& a1, float4& a2, float4& a3) {
  const float4* p0 = (const float4*)(row_ptr + koff);
  const float4* p1 = (const float4*)(row_ptr + koff + 8);
  a0 = p0[0]; a1 = p0[1]; a2 = p1[0]; a3 = p1[1];
}
__device__ __forceinline__ void cvtA(float4 a0, float4 a1, float4 a2, float4 a3,
                                     v16bf& H, v16bf& L) {
  float g[16] = {a0.x, a0.y, a0.z, a0.w, a1.x, a1.y, a1.z, a1.w,
                 a2.x, a2.y, a2.z, a2.w, a3.x, a3.y, a3.z, a3.w};
  split16(g, H, L);
}

// Async global->LDS copy of 16B (tracked by ASYNCcnt, see CDNA5 ISA)
__device__ __forceinline__ void async_copy_b128(unsigned lds_byte_off,
                                                const void* gsrc) {
  asm volatile("global_load_async_to_lds_b128 %0, %1, off" ::"v"(lds_byte_off),
               "v"((unsigned long long)(size_t)gsrc)
               : "memory");
}
__device__ __forceinline__ void wait_async0() {
  asm volatile("s_wait_asynccnt 0x0" ::: "memory");
}

// 32B B-operand read from LDS (two 16B ds loads; rows padded for 0 conflicts)
__device__ __forceinline__ v16bf loadB_lds(const unsigned short* p) {
  union { v16bf bf; float4 q[2]; } u;
  u.q[0] = *(const float4*)(p);
  u.q[1] = *(const float4*)(p + 8);
  return u.bf;
}

// ---------------------------------------------------------------------------
// Prep: x[NN][ND] fp32 -> xT hi/lo [ND][NN] bf16 (tiled transpose through LDS)
// ---------------------------------------------------------------------------
__global__ __launch_bounds__(256) void xpose_split_kernel(
    const float* __restrict__ x, unsigned short* __restrict__ xTh,
    unsigned short* __restrict__ xTl) {
  __shared__ float tile[32][33];
  const int tx = threadIdx.x, ty = threadIdx.y;  // block (32, 8)
  const int c0 = blockIdx.x * 32;
  const int r0 = blockIdx.y * 32;
#pragma unroll
  for (int i = 0; i < 32; i += 8)
    tile[ty + i][tx] = x[(size_t)(r0 + ty + i) * ND + c0 + tx];
  __syncthreads();
#pragma unroll
  for (int i = 0; i < 32; i += 8) {
    unsigned short hb, lb;
    split1(tile[tx][ty + i], hb, lb);
    size_t o = (size_t)(c0 + ty + i) * NN + (r0 + tx);
    xTh[o] = hb;
    xTl[o] = lb;
  }
}

// ---------------------------------------------------------------------------
// Prep: W[K][N] fp32 -> column-major bf16 hi/lo T[Npad][K] (zero-padded rows)
// ---------------------------------------------------------------------------
__global__ __launch_bounds__(256) void prep_w_kernel(
    const float* __restrict__ W, unsigned short* __restrict__ Th,
    unsigned short* __restrict__ Tl, int K, int N, int Npad) {
  int t = blockIdx.x * blockDim.x + threadIdx.x;
  if (t >= Npad * K) return;
  int n = t / K, k = t - n * K;
  float f = (n < N) ? W[(size_t)k * N + n] : 0.f;
  unsigned short hb, lb;
  split1(f, hb, lb);
  Th[t] = hb;
  Tl[t] = lb;
}

// ---------------------------------------------------------------------------
// Kernel 1: partial h0 = adj[:, kslab] @ x[kslab, :].
// Grid = 64 M-blocks x KSPLIT slabs; WG = 8 waves x 32 rows x 128 cols.
// adj read exactly once (1 GiB, HBM-bound); x-panels async-staged to LDS and
// shared by all 8 waves (8x cut in global B traffic vs per-wave loads).
// ---------------------------------------------------------------------------
__global__ __launch_bounds__(256) void aggr_gemm_kernel(
    const float* __restrict__ adj, const unsigned short* __restrict__ xTh,
    const unsigned short* __restrict__ xTl, float* __restrict__ h0p) {
  __shared__ __align__(32) unsigned short sPan[2][128 * PSTRIDE];
  const int tid = threadIdx.x;
  const int lane = tid & 31;
  const int wave = tid >> 5;
  const int mblk = blockIdx.x >> 2;         // 0..63
  const int s = blockIdx.x & (KSPLIT - 1);  // K slab
  const int m0 = mblk * 256 + wave * 32;
  const float* arow0 = adj + (size_t)(m0 + (lane & 15)) * NN;
  const float* arow1 = arow0 + (size_t)16 * NN;
  const int aoff = (lane & 16) ? 16 : 0;  // f32 K offset for A fetch
  const int boff = (lane & 16) ? 16 : 0;  // bf16 K offset for B fetch
  const int kbeg = s * (NN / KSPLIT);
  const int nstages = (NN / KSPLIT) / KCHUNK;

  // producer: 8 async b128 per thread per stage fills one 128x(64h+64l) panel
  auto stage_issue = [&](unsigned short* panel, int ks) {
#pragma unroll
    for (int j = 0; j < 8; ++j) {
      const int id = tid + j * 256;  // 0..2047
      const int n = id >> 4;         // panel row (x column)
      const int t16 = id & 15;       // 16 x 16B transfers per row
      const int hi = (t16 < 8) ? 1 : 0;
      const int kk = (t16 & 7) * 8;  // bf16 elements
      const unsigned short* src =
          (hi ? xTh : xTl) + (size_t)n * NN + ks + kk;
      const unsigned dst =
          (unsigned)(size_t)(panel + n * PSTRIDE + (hi ? 0 : 64) + kk);
      async_copy_b128(dst, src);
    }
  };

  v8f acc0[8], acc1[8];
#pragma unroll
  for (int t = 0; t < 8; ++t) { acc0[t] = v8f_zero(); acc1[t] = v8f_zero(); }

  stage_issue(sPan[0], kbeg);

  for (int st = 0; st < nstages; ++st) {
    wait_async0();
    __syncthreads();  // panel[st&1] fully written; prior reads finished
    const int buf = st & 1;
    if (st + 1 < nstages)
      stage_issue(sPan[buf ^ 1], kbeg + (st + 1) * KCHUNK);  // overlap w/ compute
    const int ks = kbeg + st * KCHUNK;
#pragma unroll
    for (int kq = 0; kq < 2; ++kq) {
      const int k0 = ks + kq * 32;
      if (k0 + 1024 < kbeg + NN / KSPLIT)
        __builtin_prefetch(arow0 + k0 + 1024, 0, 1);  // global_prefetch_b8
      float4 a0, a1, a2, a3;
      v16bf Ah0, Al0, Ah1, Al1;
      fetchA(arow0, k0 + aoff, a0, a1, a2, a3);
      cvtA(a0, a1, a2, a3, Ah0, Al0);
      fetchA(arow1, k0 + aoff, a0, a1, a2, a3);
      cvtA(a0, a1, a2, a3, Ah1, Al1);
#pragma unroll
      for (int nt = 0; nt < 8; ++nt) {
        const unsigned short* rowp =
            &sPan[buf][(nt * 16 + (lane & 15)) * PSTRIDE];
        v16bf Bh = loadB_lds(rowp + kq * 32 + boff);
        v16bf Bl = loadB_lds(rowp + 64 + kq * 32 + boff);
        acc0[nt] = wmma_bf16(Ah0, Bh, acc0[nt]);
        acc0[nt] = wmma_bf16(Al0, Bh, acc0[nt]);
        acc0[nt] = wmma_bf16(Ah0, Bl, acc0[nt]);
        acc1[nt] = wmma_bf16(Ah1, Bh, acc1[nt]);
        acc1[nt] = wmma_bf16(Al1, Bh, acc1[nt]);
        acc1[nt] = wmma_bf16(Ah1, Bl, acc1[nt]);
      }
    }
    __syncthreads();  // all waves done reading panel[buf] before it is refilled
  }

  // C layout: VGPR r -> row (r + 8*(lane>=16)), col = lane&15
  float* h0s = h0p + (size_t)s * ((size_t)NN * ND);
  const int mb = (lane & 16) ? 8 : 0;
#pragma unroll
  for (int nt = 0; nt < 8; ++nt) {
    const int n = nt * 16 + (lane & 15);
#pragma unroll
    for (int r = 0; r < 8; ++r) {
      h0s[(size_t)(m0 + mb + r) * ND + n] = acc0[nt][r];
      h0s[(size_t)(m0 + 16 + mb + r) * ND + n] = acc1[nt][r];
    }
  }
}

// ---------------------------------------------------------------------------
// Fused post-pipeline helpers
// ---------------------------------------------------------------------------
__device__ __forceinline__ v8f gemm_tile(const float* sA, int astride,
                                         const unsigned short* BTh,
                                         const unsigned short* BTl, int K, int n0,
                                         int lane) {
  v8f acc = v8f_zero();
  const int n = n0 + (lane & 15);
  const int aoff = (lane & 16) ? 16 : 0;
  const int boff = (lane & 16) ? 16 : 0;
  const float* arow = sA + (lane & 15) * astride;
  for (int k0 = 0; k0 < K; k0 += 32) {
    float4 a0, a1, a2, a3;
    fetchA(arow, k0 + aoff, a0, a1, a2, a3);
    v16bf Ah, Al;
    cvtA(a0, a1, a2, a3, Ah, Al);
    v16bf Bh = *(const v16bf*)(BTh + (size_t)n * K + k0 + boff);
    v16bf Bl = *(const v16bf*)(BTl + (size_t)n * K + k0 + boff);
    acc = wmma_bf16(Ah, Bh, acc);
    acc = wmma_bf16(Al, Bh, acc);
    acc = wmma_bf16(Ah, Bl, acc);
  }
  return acc;
}

__device__ __forceinline__ void storeC(float* s, int stride, v8f c, int n0, int lane,
                                       bool relu) {
  const int n = n0 + (lane & 15);
  const int mb = (lane & 16) ? 8 : 0;
#pragma unroll
  for (int r = 0; r < 8; ++r) {
    float v = c[r];
    if (relu) v = v > 0.f ? v : 0.f;
    s[(mb + r) * stride + n] = v;
  }
}

__device__ __forceinline__ void ln_row(const float* in, float* outp,
                                       const float* __restrict__ scale) {
  float s = 0.f;
  for (int i = 0; i < ND; ++i) s += in[i];
  const float mean = s * (1.0f / ND);
  float v = 0.f;
  for (int i = 0; i < ND; ++i) {
    float d = in[i] - mean;
    v += d * d;
  }
  v *= (1.0f / ND);
  const float r = rsqrtf(v + 1e-6f);
  for (int i = 0; i < ND; ++i) outp[i] = (in[i] - mean) * r * scale[i];
}

// ---------------------------------------------------------------------------
// Kernel 2: one 16-row tile per 256-thread WG (8 waves, one N-tile per wave).
// ---------------------------------------------------------------------------
__global__ __launch_bounds__(256) void post_kernel(
    const float* __restrict__ h0p,
    const unsigned short* __restrict__ WqTh, const unsigned short* __restrict__ WqTl,
    const unsigned short* __restrict__ WkTh, const unsigned short* __restrict__ WkTl,
    const unsigned short* __restrict__ WvTh, const unsigned short* __restrict__ WvTl,
    const unsigned short* __restrict__ WoTh, const unsigned short* __restrict__ WoTl,
    const unsigned short* __restrict__ W1Th, const unsigned short* __restrict__ W1Tl,
    const unsigned short* __restrict__ W2Th, const unsigned short* __restrict__ W2Tl,
    const unsigned short* __restrict__ WcTh, const unsigned short* __restrict__ WcTl,
    const float* __restrict__ ln1s, const float* __restrict__ ln2s,
    float* __restrict__ out) {
  __shared__ __align__(16) float sA[16 * 128];  // h0 / normalized h (residuals)
  __shared__ __align__(16) float sB[16 * 384];  // async staging, qkv, pre-LN
  __shared__ __align__(16) float sC[16 * 256];  // async staging, attn, ffn hidden
  const int tid = threadIdx.x, lane = tid & 31, w = tid >> 5;
  const int r0 = blockIdx.x * 16;

  // --- stage the 4 partial-h0 tiles via async LDS loads, reduce into sA ---
  {
#pragma unroll
    for (int s = 0; s < KSPLIT; ++s) {
      const float* src = h0p + (size_t)s * ((size_t)NN * ND) + (size_t)r0 * ND;
      float* dst = (s < 3) ? &sB[s * 2048] : &sC[0];
#pragma unroll
      for (int j = 0; j < 2; ++j) {
        const int e = (tid + j * 256) * 4;  // 16B-granular element index
        async_copy_b128((unsigned)(size_t)(dst + e), src + e);
      }
    }
    wait_async0();
    __syncthreads();
    for (int i = tid; i < 16 * 128; i += 256)
      sA[i] = sB[i] + sB[2048 + i] + sB[4096 + i] + sC[i];
  }
  __syncthreads();

  // --- QKV projections: wave w owns N-slice [w*16, w*16+16) of each ---
  {
    v8f cq = gemm_tile(sA, 128, WqTh, WqTl, 128, w * 16, lane);
    v8f ck = gemm_tile(sA, 128, WkTh, WkTl, 128, w * 16, lane);
    v8f cv = gemm_tile(sA, 128, WvTh, WvTl, 128, w * 16, lane);
    storeC(sB, 384, cq, w * 16, lane, false);
    storeC(sB, 384, ck, 128 + w * 16, lane, false);
    storeC(sB, 384, cv, 256 + w * 16, lane, false);
  }
  __syncthreads();

  // --- elementwise q*k scores, softmax over the 8 heads, * v -> attn ---
  {
    const int row = tid >> 4, d = tid & 15;  // 256 threads = 16 rows x 16 dims
    const float* q = &sB[row * 384];
    const float* kk = q + 128;
    const float* vv = q + 256;
    float sc[8];
    float mx = -3.0e38f;
#pragma unroll
    for (int h = 0; h < 8; ++h) {
      sc[h] = q[d * 8 + h] * kk[d * 8 + h] * 0.25f;  // 1/sqrt(HEAD_DIM=16)
      mx = fmaxf(mx, sc[h]);
    }
    float sum = 0.f;
#pragma unroll
    for (int h = 0; h < 8; ++h) {
      sc[h] = expf(sc[h] - mx);
      sum += sc[h];
    }
    const float inv = 1.f / sum;
#pragma unroll
    for (int h = 0; h < 8; ++h) sC[row * 256 + d * 8 + h] = sc[h] * inv * vv[d * 8 + h];
  }
  __syncthreads();

  // --- attn @ Wo + residual(h0) -> pre-LN in sB cols 0..127 ---
  {
    v8f c = gemm_tile(sC, 256, WoTh, WoTl, 128, w * 16, lane);
    const int n = w * 16 + (lane & 15);
    const int mb = (lane & 16) ? 8 : 0;
#pragma unroll
    for (int r = 0; r < 8; ++r)
      sB[(mb + r) * 384 + n] = c[r] + sA[(mb + r) * 128 + n];
  }
  __syncthreads();
  if (tid < 16) ln_row(&sB[tid * 384], &sA[tid * 128], ln1s);  // sA <- h (post-LN1)
  __syncthreads();

  // --- FFN up (N=256) + relu ---
  {
    v8f c0 = gemm_tile(sA, 128, W1Th, W1Tl, 128, w * 16, lane);
    v8f c1 = gemm_tile(sA, 128, W1Th, W1Tl, 128, 128 + w * 16, lane);
    storeC(sC, 256, c0, w * 16, lane, true);
    storeC(sC, 256, c1, 128 + w * 16, lane, true);
  }
  __syncthreads();

  // --- FFN down (K=256) + residual(h) -> pre-LN in sB ---
  {
    v8f c = gemm_tile(sC, 256, W2Th, W2Tl, 256, w * 16, lane);
    const int n = w * 16 + (lane & 15);
    const int mb = (lane & 16) ? 8 : 0;
#pragma unroll
    for (int r = 0; r < 8; ++r)
      sB[(mb + r) * 384 + n] = c[r] + sA[(mb + r) * 128 + n];
  }
  __syncthreads();
  if (tid < 16) ln_row(&sB[tid * 384], &sA[tid * 128], ln2s);  // sA <- h2
  __syncthreads();

  // --- classifier: 40 cols padded to 48 (3 tiles, waves 0..2; wave-uniform) ---
  if (w < 3) {
    v8f c = gemm_tile(sA, 128, WcTh, WcTl, 128, w * 16, lane);
    const int n = w * 16 + (lane & 15);
    const int mb = (lane & 16) ? 8 : 0;
    if (n < 40) {
#pragma unroll
      for (int r = 0; r < 8; ++r) out[(size_t)(r0 + mb + r) * 40 + n] = c[r];
    }
  }
}

// ---------------------------------------------------------------------------
extern "C" void kernel_launch(void* const* d_in, const int* in_sizes, int n_in,
                              void* d_out, int out_size, void* d_ws, size_t ws_size,
                              hipStream_t stream) {
  const float* x    = (const float*)d_in[0];
  const float* adj  = (const float*)d_in[1];
  const float* Wq   = (const float*)d_in[2];
  const float* Wk   = (const float*)d_in[3];
  const float* Wv   = (const float*)d_in[4];
  const float* Wo   = (const float*)d_in[5];
  const float* ln1  = (const float*)d_in[6];
  const float* W1   = (const float*)d_in[7];
  const float* W2   = (const float*)d_in[8];
  const float* ln2  = (const float*)d_in[9];
  const float* Wout = (const float*)d_in[10];
  float* out = (float*)d_out;

  char* wsp = (char*)d_ws;
  size_t off = 0;
  auto alloc = [&](size_t bytes) -> void* {
    void* p = wsp + off;
    off = (off + bytes + 255) & ~(size_t)255;
    return p;
  };
  unsigned short* xTh = (unsigned short*)alloc((size_t)ND * NN * 2);
  unsigned short* xTl = (unsigned short*)alloc((size_t)ND * NN * 2);
  float* h0p = (float*)alloc((size_t)KSPLIT * NN * ND * 4);
  unsigned short* WqTh = (unsigned short*)alloc(128 * 128 * 2);
  unsigned short* WqTl = (unsigned short*)alloc(128 * 128 * 2);
  unsigned short* WkTh = (unsigned short*)alloc(128 * 128 * 2);
  unsigned short* WkTl = (unsigned short*)alloc(128 * 128 * 2);
  unsigned short* WvTh = (unsigned short*)alloc(128 * 128 * 2);
  unsigned short* WvTl = (unsigned short*)alloc(128 * 128 * 2);
  unsigned short* WoTh = (unsigned short*)alloc(128 * 128 * 2);
  unsigned short* WoTl = (unsigned short*)alloc(128 * 128 * 2);
  unsigned short* W1Th = (unsigned short*)alloc(256 * 128 * 2);
  unsigned short* W1Tl = (unsigned short*)alloc(256 * 128 * 2);
  unsigned short* W2Th = (unsigned short*)alloc(128 * 256 * 2);
  unsigned short* W2Tl = (unsigned short*)alloc(128 * 256 * 2);
  unsigned short* WcTh = (unsigned short*)alloc(48 * 128 * 2);
  unsigned short* WcTl = (unsigned short*)alloc(48 * 128 * 2);

  // prep: transpose/split x; split weights to column-major bf16 hi/lo
  dim3 tb(32, 8);
  xpose_split_kernel<<<dim3(ND / 32, NN / 32), tb, 0, stream>>>(x, xTh, xTl);
  prep_w_kernel<<<64, 256, 0, stream>>>(Wq, WqTh, WqTl, 128, 128, 128);
  prep_w_kernel<<<64, 256, 0, stream>>>(Wk, WkTh, WkTl, 128, 128, 128);
  prep_w_kernel<<<64, 256, 0, stream>>>(Wv, WvTh, WvTl, 128, 128, 128);
  prep_w_kernel<<<64, 256, 0, stream>>>(Wo, WoTh, WoTl, 128, 128, 128);
  prep_w_kernel<<<128, 256, 0, stream>>>(W1, W1Th, W1Tl, 128, 256, 256);
  prep_w_kernel<<<128, 256, 0, stream>>>(W2, W2Th, W2Tl, 256, 128, 128);
  prep_w_kernel<<<24, 256, 0, stream>>>(Wout, WcTh, WcTl, 128, 40, 48);

  // big aggregation GEMM: partial h0 per K-slab (LDS-staged B panels)
  aggr_gemm_kernel<<<(NN / 256) * KSPLIT, 256, 0, stream>>>(adj, xTh, xTl, h0p);

  // fused transformer block + classifier (reduces the K-slabs on load)
  post_kernel<<<NN / 16, 256, 0, stream>>>(h0p, WqTh, WqTl, WkTh, WkTl, WvTh, WvTl,
                                           WoTh, WoTl, W1Th, W1Tl, W2Th, W2Tl, WcTh,
                                           WcTl, ln1, ln2, out);
}
